// QLSTM_2bit_44633300140401
// MI455X (gfx1250) — 8-bit matrix operand order repaired and checked by reading, compiled, not run on hardware
//
#include <hip/hip_runtime.h>

typedef __attribute__((ext_vector_type(8))) int v8i;

#define B_    64
#define T_    512
#define E_    512
#define H_    512
#define FOURH 2048
#define V_    32000
#define TS    0.1f

__device__ __forceinline__ float tern_val(float x) {
    float r = rintf(x / TS);
    r = fminf(1.0f, fmaxf(-1.0f, r));
    return r * TS;
}
__device__ __forceinline__ int tern_code(float x) {
    float r = rintf(x / TS);
    r = fminf(1.0f, fmaxf(-1.0f, r));
    return (int)r;
}
__device__ __forceinline__ float sigmoidf(float x) {
    return 1.0f / (1.0f + expf(-x));
}

__global__ void qlstm_maxabs_part(const float* __restrict__ w, int n,
                                  float* __restrict__ partial) {
    __shared__ float sm[256];
    float m = 0.0f;
    for (int i = blockIdx.x * 256 + threadIdx.x; i < n; i += gridDim.x * 256)
        m = fmaxf(m, fabsf(w[i]));
    sm[threadIdx.x] = m;
    __syncthreads();
    for (int s = 128; s > 0; s >>= 1) {
        if (threadIdx.x < s) sm[threadIdx.x] = fmaxf(sm[threadIdx.x], sm[threadIdx.x + s]);
        __syncthreads();
    }
    if (threadIdx.x == 0) partial[blockIdx.x] = sm[0];
}

__global__ void qlstm_maxabs_final(const float* __restrict__ partial, int n,
                                   float* __restrict__ scale) {
    __shared__ float sm[256];
    float m = 0.0f;
    for (int i = threadIdx.x; i < n; i += 256) m = fmaxf(m, partial[i]);
    sm[threadIdx.x] = m;
    __syncthreads();
    for (int s = 128; s > 0; s >>= 1) {
        if (threadIdx.x < s) sm[threadIdx.x] = fmaxf(sm[threadIdx.x], sm[threadIdx.x + s]);
        __syncthreads();
    }
    if (threadIdx.x == 0) *scale = exp2f(ceilf(log2f(sm[0] / 127.0f)));
}

__global__ void qlstm_tern_emb(const float* __restrict__ emb,
                               const float* __restrict__ scale_ptr,
                               signed char* __restrict__ eq, int n) {
    int i = blockIdx.x * 256 + threadIdx.x;
    if (i >= n) return;
    float s = scale_ptr[0];
    float inv_s = 1.0f / s;
    float q = rintf(emb[i] * inv_s);
    q = fminf(127.0f, fmaxf(-128.0f, q)) * s;
    eq[i] = (signed char)tern_code(q);
}

__global__ void qlstm_pack_wfrag(const float* __restrict__ wih,
                                 const float* __restrict__ whh,
                                 signed char* __restrict__ fih,
                                 signed char* __restrict__ fhh) {
    int id = blockIdx.x * 256 + threadIdx.x;
    int which = id >> 20;
    int r = id & ((1 << 20) - 1);
    int byteidx = r & 31;
    int lane    = (r >> 5) & 31;
    int kt      = (r >> 10) & 7;
    int nt      = r >> 13;
    int v  = byteidx >> 2;
    int bb = byteidx & 3;
    int kbase  = (lane >> 4) << 4;
    int klocal = (v < 4) ? (kbase + v * 4 + bb) : (32 + kbase + (v - 4) * 4 + bb);
    int k = (kt << 6) + klocal;
    int n = (nt << 4) + (lane & 15);
    const float* w = which ? whh : wih;
    signed char code = (signed char)tern_code(w[n * 512 + k]);
    (which ? fhh : fih)[r] = code;
}

__global__ __launch_bounds__(128) void qlstm_step_gemm(
    const int* __restrict__ text,
    const signed char* __restrict__ eq,
    const signed char* __restrict__ wfrag_ih,
    const signed char* __restrict__ wfrag_hh,
    const signed char* __restrict__ hq,
    int* __restrict__ acc_out, int t) {

    __shared__ __align__(32) unsigned long long a_lds[4][32][4];

    const int tid  = threadIdx.x;
    const int lane = tid & 31;
    const int wave = tid >> 5;
    const int nt   = blockIdx.x * 4 + wave;
    const int b_fill = (wave << 4) + (lane & 15);
    const int kb     = (lane >> 4) << 4;

    const int token = text[b_fill * T_ + t];
    const signed char* eqrow = eq + (size_t)token * E_;
    const signed char* hrow  = hq + b_fill * H_;

    v8i acc[4] = {};

    for (int phase = 0; phase < 2; ++phase) {
        const signed char* wf   = phase ? wfrag_hh : wfrag_ih;
        const signed char* arow = phase ? hrow : eqrow;
        for (int kt = 0; kt < 8; ++kt) {
            #pragma unroll
            for (int cch = 0; cch < 4; ++cch) {
                int kk = (kt << 6) + kb + ((cch & 1) << 3) + ((cch >> 1) << 5);
                a_lds[wave][lane][cch] =
                    *(const unsigned long long*)(arow + kk);
            }
            __syncthreads();

            const signed char* bp =
                wf + ((((size_t)nt * 8 + kt) * 32 + lane) << 5);
            int4 bA = *(const int4*)bp;
            int4 bB = *(const int4*)(bp + 16);
            v8i bf = {bA.x, bA.y, bA.z, bA.w, bB.x, bB.y, bB.z, bB.w};
            if (kt < 7)
                __builtin_prefetch(bp + 1024, 0, 3);

            #pragma unroll
            for (int mt = 0; mt < 4; ++mt) {
                v8i af = *(const v8i*)(&a_lds[mt][lane][0]);
                acc[mt] = __builtin_amdgcn_wmma_i32_16x16x64_iu8(
                     true, af,  true, bf, acc[mt],
                     false,  false);
            }
            __syncthreads();
        }
    }

    const int n  = (nt << 4) + (lane & 15);
    const int mb = (lane >> 4) << 3;
    for (int mt = 0; mt < 4; ++mt) {
        int mrow = (mt << 4) + mb;
        #pragma unroll
        for (int r = 0; r < 8; ++r)
            acc_out[(mrow + r) * FOURH + n] = acc[mt][r];
    }
}

__global__ void qlstm_step_point(const int* __restrict__ acc,
                                 const float* __restrict__ bias,
                                 signed char* __restrict__ hq,
                                 float* __restrict__ c) {
    int idx = blockIdx.x * blockDim.x + threadIdx.x;
    int b = idx >> 9;
    int j = idx & 511;
    const int* row = acc + b * FOURH;
    float ai = tern_val(0.01f * (float)row[j]          + bias[j]);
    float af = tern_val(0.01f * (float)row[j + 512]    + bias[j + 512]);
    float ag = tern_val(0.01f * (float)row[j + 1024]   + bias[j + 1024]);
    float ao = tern_val(0.01f * (float)row[j + 1536]   + bias[j + 1536]);
    float iv = tern_val(sigmoidf(ai));
    float fv = tern_val(sigmoidf(af));
    float ov = tern_val(sigmoidf(ao));
    float gv = tern_val(tanhf(ag));
    float cn = fv * c[idx] + iv * gv;
    c[idx] = cn;
    float hv = tern_val(ov * tern_val(tanhf(cn)));
    hq[idx] = (signed char)(int)rintf(hv / TS);
}

__global__ void qlstm_fc(const signed char* __restrict__ hq,
                         const float* __restrict__ fcw,
                         float* __restrict__ out) {
    int idx = threadIdx.x;
    int b = idx >> 1, o = idx & 1;
    int s = 0;
    for (int j = 0; j < 512; ++j)
        s += tern_code(fcw[o * 512 + j]) * (int)hq[b * 512 + j];
    out[b * 2 + o] = 0.01f * (float)s;
}

extern "C" void kernel_launch(void* const* d_in, const int* in_sizes, int n_in,
                              void* d_out, int out_size, void* d_ws, size_t ws_size,
                              hipStream_t stream) {
    (void)in_sizes; (void)n_in; (void)out_size; (void)ws_size;
    const int*   text = (const int*)d_in[0];
    const float* emb  = (const float*)d_in[2];
    const float* wih  = (const float*)d_in[3];
    const float* whh  = (const float*)d_in[4];
    const float* bias = (const float*)d_in[5];
    const float* fcw  = (const float*)d_in[6];
    float* out = (float*)d_out;

    const int EQN = V_ * E_;
    signed char* fih  = (signed char*)d_ws;
    signed char* fhh  = fih + (1 << 20);
    signed char* eqt  = fhh + (1 << 20);
    signed char* hq   = eqt + EQN;
    float* cbuf       = (float*)(hq + B_ * H_);
    int*   accb       = (int*)((char*)cbuf + (size_t)B_ * H_ * 4);
    float* partial    = (float*)((char*)accb + (size_t)B_ * FOURH * 4);
    float* scaleb     = partial + 256;

    hipMemsetAsync(hq,   0, B_ * H_,              stream);
    hipMemsetAsync(cbuf, 0, (size_t)B_ * H_ * 4,  stream);

    qlstm_maxabs_part<<<256, 256, 0, stream>>>(emb, EQN, partial);
    qlstm_maxabs_final<<<1, 256, 0, stream>>>(partial, 256, scaleb);
    qlstm_tern_emb<<<EQN / 256, 256, 0, stream>>>(emb, scaleb, eqt, EQN);
    qlstm_pack_wfrag<<<(2 * (1 << 20)) / 256, 256, 0, stream>>>(wih, whh, fih, fhh);

    for (int t = 0; t < T_; ++t) {
        qlstm_step_gemm<<<32, 128, 0, stream>>>(text, eqt, fih, fhh, hq,
                                                accb, t);
        qlstm_step_point<<<(B_ * H_) / 256, 256, 0, stream>>>(accb, bias, hq, cbuf);
    }
    qlstm_fc<<<1, 128, 0, stream>>>(hq, fcw, out);
}
